// BaseMultiHeadTEAttention_49374944034918
// MI455X (gfx1250) — matrix operand order repaired and checked by reading, not compiled, not run on hardware
//
#include <hip/hip_runtime.h>
#include <hip/hip_bf16.h>
#include <math.h>

// ---------------- common types / helpers ----------------

typedef __attribute__((ext_vector_type(16))) __bf16       v16bf;
typedef __attribute__((ext_vector_type(8)))  float        v8f;
typedef __attribute__((ext_vector_type(8)))  unsigned int v8u;

__device__ __forceinline__ v16bf asbf(v8u u) {
    return __builtin_bit_cast(v16bf, u);
}

__device__ __forceinline__ unsigned short f2bf(float x) {
    unsigned int u = __builtin_bit_cast(unsigned int, x);
    unsigned int r = (u + 0x7FFFu + ((u >> 16) & 1u)) >> 16;   // RNE
    return (unsigned short)r;
}
__device__ __forceinline__ float bf2f(unsigned short h) {
    unsigned int u = ((unsigned int)h) << 16;
    return __builtin_bit_cast(float, u);
}
__device__ __forceinline__ unsigned int pack2(unsigned short lo, unsigned short hi) {
    return (unsigned int)lo | ((unsigned int)hi << 16);
}

// A-matrix 16x32 bf16 VGPR layout (K offset for vgpr r, lane half).
__device__ __forceinline__ int a_kk(int r, int half) {
    return (r < 4) ? (half * 8 + 2 * r) : (16 + half * 8 + 2 * (r - 4));
}
// B-matrix 32x16 bf16 VGPR layout (same K order as A).
__device__ __forceinline__ int b_kk(int r, int half) {
    return a_kk(r, half);
}

__device__ __forceinline__ v8f wmma_bf16(v8u a, v8u b, v8f c) {
    return __builtin_amdgcn_wmma_f32_16x16x32_bf16(
        /*neg_a=*/false, asbf(a), /*neg_b=*/false, asbf(b),
        /*c_mod=*/(short)0, c, /*reuse_a=*/false, /*reuse_b=*/false);
}

#define MB   8
#define NQC  1024
#define NKC  1024
#define HH   8
#define HDD  32

// ---------------- kernel 1a: projection GEMM -> bf16, row-major ----------------

__global__ void proj_gemm_n(const float* __restrict__ Z, const float* __restrict__ W,
                            unsigned short* __restrict__ dstN) {
    int gid  = blockIdx.x * (blockDim.x >> 5) + (threadIdx.x >> 5);
    int lane = threadIdx.x & 31;
    int half = lane >> 4;
    int ln   = lane & 15;
    int row0 = (gid >> 4) * 16;
    int col0 = (gid & 15) * 16;

    v8f acc = {};
    for (int kk0 = 0; kk0 < 256; kk0 += 32) {
        v8u Au = {}, Bu = {};
#pragma unroll
        for (int r = 0; r < 8; ++r) {
            const float* pz = Z + (size_t)(row0 + ln) * 256 + kk0 + a_kk(r, half);
            Au[r] = pack2(f2bf(pz[0]), f2bf(pz[1]));
            int kb = kk0 + b_kk(r, half);
            Bu[r] = pack2(f2bf(W[(size_t)kb * 256 + col0 + ln]),
                          f2bf(W[(size_t)(kb + 1) * 256 + col0 + ln]));
        }
        acc = wmma_bf16(Au, Bu, acc);
    }
#pragma unroll
    for (int r = 0; r < 8; ++r)
        dstN[(size_t)(row0 + r + 8 * half) * 256 + col0 + ln] = f2bf(acc[r]);
}

// ---------------- kernel 1b: projection GEMM -> bf16, transposed [m][inner][key] ----------------

__global__ void proj_gemm_t(const float* __restrict__ Z, const float* __restrict__ W,
                            unsigned short* __restrict__ dstT) {
    int gid  = blockIdx.x * (blockDim.x >> 5) + (threadIdx.x >> 5);
    int lane = threadIdx.x & 31;
    int half = lane >> 4;
    int ln   = lane & 15;
    int row0 = (gid >> 4) * 16;
    int col0 = (gid & 15) * 16;

    v8f acc = {};
    for (int kk0 = 0; kk0 < 256; kk0 += 32) {
        v8u Au = {}, Bu = {};
#pragma unroll
        for (int r = 0; r < 8; ++r) {
            const float* pz = Z + (size_t)(row0 + ln) * 256 + kk0 + a_kk(r, half);
            Au[r] = pack2(f2bf(pz[0]), f2bf(pz[1]));
            int kb = kk0 + b_kk(r, half);
            Bu[r] = pack2(f2bf(W[(size_t)kb * 256 + col0 + ln]),
                          f2bf(W[(size_t)(kb + 1) * 256 + col0 + ln]));
        }
        acc = wmma_bf16(Au, Bu, acc);
    }
#pragma unroll
    for (int r = 0; r < 8; ++r) {
        int grow = row0 + r + 8 * half;
        int m = grow >> 10, k = grow & 1023;
        dstT[((size_t)m * 256 + col0 + ln) * 1024 + k] = f2bf(acc[r]);
    }
}

// ---------------- kernel 2: fused attention (all three GEMM stages on WMMA) ----------------
// Grid: M * NQ/32 blocks, 64 threads (2 waves); each wave owns a 16-query tile.

__global__ void __launch_bounds__(64)
attn_kernel(const unsigned short* __restrict__ Qb,
            const unsigned short* __restrict__ Kb,
            const unsigned short* __restrict__ Vt,
            const float* __restrict__ xq,  const float* __restrict__ xkv,
            const float* __restrict__ W1,  const float* __restrict__ b1,
            const float* __restrict__ W2,  const float* __restrict__ b2,
            float* __restrict__ O) {
    // per-wave LDS tiles
    __shared__ __align__(16) unsigned short Din[2][512 * 16]; // MLP input [pos][diff0,diff1,td0..7,0..0]
    __shared__ __align__(16) unsigned short Dls[2][512 * 8];  // dots -> probs [pos][h]
    __shared__ __align__(16) unsigned short Hst[2][16 * 32];  // layer-1 transpose staging
    __shared__ float Als[2][16][8];                           // row broadcast (alpha / 1/l)

    int wv   = threadIdx.x >> 5;
    int lane = threadIdx.x & 31;
    int half = lane >> 4;
    int ln   = lane & 15;
    int m    = blockIdx.x >> 5;
    int q0   = (blockIdx.x & 31) * 32 + wv * 16;

    const float scale = 0.17677669529663687f; // 32^-0.5

    // ---- constant weight fragments (registers) ----
    v8u W1f[2]; v8u W2f = {};
#pragma unroll
    for (int nt = 0; nt < 2; ++nt) {
#pragma unroll
        for (int r = 0; r < 8; ++r) {
            unsigned int v = 0;
            int kw = a_kk(r, half);
            if (kw < 10) {
                int col = nt * 16 + ln;
                unsigned short lo = f2bf(W1[(size_t)(kw) * 32 + col]);
                unsigned short hi = (kw + 1 < 10)
                                        ? f2bf(W1[(size_t)(kw + 1) * 32 + col])
                                        : (unsigned short)0;
                v = pack2(lo, hi);
            }
            W1f[nt][r] = v;
        }
    }
#pragma unroll
    for (int r = 0; r < 8; ++r) {
        int kh = a_kk(r, half);
        unsigned int v = 0;
        if (ln < 8) v = pack2(f2bf(W2[(size_t)kh * 8 + ln]),
                              f2bf(W2[(size_t)(kh + 1) * 8 + ln]));
        W2f[r] = v;
    }
    float bias1[2];
    bias1[0] = b1[ln];
    bias1[1] = b1[16 + ln];
    float bias2 = (ln < 8) ? b2[ln] : 0.0f;

    // ---- flash accumulators ----
    v8f acc[HH][2];
#pragma unroll
    for (int h = 0; h < HH; ++h) { acc[h][0] = v8f{}; acc[h][1] = v8f{}; }
    float m_run[HH], l_run[HH];
#pragma unroll
    for (int h = 0; h < HH; ++h) { m_run[h] = -1e30f; l_run[h] = 0.0f; }

    int qown = ln; // row ownership for softmax: keys [half*16, half*16+16)
    float xq0 = xq[((size_t)m * NQC + q0 + qown) * 2 + 0];
    float xq1 = xq[((size_t)m * NQC + q0 + qown) * 2 + 1];

    // zero the pad columns [10..15] of Din once for owned positions
#pragma unroll
    for (int j = 0; j < 16; ++j) {
        int pos = qown * 32 + half * 16 + j;
        unsigned int* p = (unsigned int*)&Din[wv][pos * 16 + 10];
        p[0] = 0; p[1] = 0; p[2] = 0;
    }

    for (int kb = 0; kb < NKC / 32; ++kb) {
        int kbase = kb * 32;

        // position diffs for owned positions -> Din[pos][0..1]
#pragma unroll
        for (int j = 0; j < 16; ++j) {
            int kk = half * 16 + j;
            int kg = kbase + kk;
            float d0 = xq0 - xkv[((size_t)m * NKC + kg) * 2 + 0];
            float d1 = xq1 - xkv[((size_t)m * NKC + kg) * 2 + 1];
            *(unsigned int*)&Din[wv][(qown * 32 + kk) * 16] = pack2(f2bf(d0), f2bf(d1));
        }

        // ---- stage 1: token dots S = (Q K^T)*scale -> Din[pos][2+h] ----
        const unsigned short* qrow = Qb + ((size_t)m * NQC + q0 + ln) * 256;
#pragma unroll
        for (int h = 0; h < HH; ++h) {
            v8u Au = {};
#pragma unroll
            for (int r = 0; r < 8; ++r)
                Au[r] = *(const unsigned int*)(qrow + h * 32 + a_kk(r, half));
            for (int sub = 0; sub < 2; ++sub) {
                const unsigned short* krow =
                    Kb + ((size_t)m * NKC + kbase + sub * 16 + ln) * 256;
                v8u Bu = {};
#pragma unroll
                for (int r = 0; r < 8; ++r)
                    Bu[r] = *(const unsigned int*)(krow + h * 32 + b_kk(r, half));
                v8f s = wmma_bf16(Au, Bu, v8f{});
#pragma unroll
                for (int r = 0; r < 8; ++r) {
                    int pos = (r + 8 * half) * 32 + sub * 16 + ln;
                    Din[wv][pos * 16 + 2 + h] = f2bf(s[r] * scale);
                }
            }
        }
        // prefetch next key block (K rows and Vt columns) -> global_prefetch_b8
        if (kb + 1 < NKC / 32) {
            __builtin_prefetch(Kb + ((size_t)m * NKC + kbase + 32 + lane) * 256, 0, 3);
            __builtin_prefetch(Vt + ((size_t)m * 256 + lane * 8) * 1024 + kbase + 32, 0, 3);
        }
        __syncthreads();

        // ---- stage 2: kernel MLP as WMMA over 32 tiles of 16 positions ----
        for (int pt = 0; pt < 32; ++pt) {
            int pos0 = pt * 16;
            v8u Au = {};                       // K=16 real (cols 10..15 zero), K=16..31 zero
#pragma unroll
            for (int r = 0; r < 4; ++r)
                Au[r] = *(const unsigned int*)&Din[wv][(pos0 + ln) * 16 + half * 8 + 2 * r];
#pragma unroll
            for (int nt = 0; nt < 2; ++nt) {
                v8f c;
#pragma unroll
                for (int r = 0; r < 8; ++r) c[r] = bias1[nt];
                v8f hsum = wmma_bf16(Au, W1f[nt], c);
#pragma unroll
                for (int r = 0; r < 8; ++r)
                    Hst[wv][(r + 8 * half) * 32 + nt * 16 + ln] =
                        f2bf(fmaxf(hsum[r], 0.0f));
            }
            __syncthreads();
            v8u Au2 = {};
#pragma unroll
            for (int r = 0; r < 8; ++r)
                Au2[r] = *(const unsigned int*)&Hst[wv][ln * 32 + a_kk(r, half)];
            v8f c2;
#pragma unroll
            for (int r = 0; r < 8; ++r) c2[r] = bias2;
            v8f dt = wmma_bf16(Au2, W2f, c2);
            if (ln < 8) {
#pragma unroll
                for (int r = 0; r < 8; ++r)
                    Dls[wv][(pos0 + r + 8 * half) * 8 + ln] = f2bf(dt[r]);
            }
            __syncthreads();
        }

        // ---- stage 2b: online softmax over the 32-key block ----
        float mblk[HH];
#pragma unroll
        for (int h = 0; h < HH; ++h) mblk[h] = -1e30f;
#pragma unroll
        for (int j = 0; j < 16; ++j) {
            int pos = qown * 32 + half * 16 + j;
            uint4 w = *(const uint4*)&Dls[wv][pos * 8];
            mblk[0] = fmaxf(mblk[0], bf2f((unsigned short)(w.x & 0xffff)));
            mblk[1] = fmaxf(mblk[1], bf2f((unsigned short)(w.x >> 16)));
            mblk[2] = fmaxf(mblk[2], bf2f((unsigned short)(w.y & 0xffff)));
            mblk[3] = fmaxf(mblk[3], bf2f((unsigned short)(w.y >> 16)));
            mblk[4] = fmaxf(mblk[4], bf2f((unsigned short)(w.z & 0xffff)));
            mblk[5] = fmaxf(mblk[5], bf2f((unsigned short)(w.z >> 16)));
            mblk[6] = fmaxf(mblk[6], bf2f((unsigned short)(w.w & 0xffff)));
            mblk[7] = fmaxf(mblk[7], bf2f((unsigned short)(w.w >> 16)));
        }
        float mnew[HH], alpha[HH], lblk[HH];
#pragma unroll
        for (int h = 0; h < HH; ++h) {
            mblk[h]  = fmaxf(mblk[h], __shfl_xor(mblk[h], 16, 32));
            mnew[h]  = fmaxf(m_run[h], mblk[h]);
            alpha[h] = __expf(m_run[h] - mnew[h]);
            lblk[h]  = 0.0f;
        }
        for (int j = 0; j < 16; ++j) {
            int pos = qown * 32 + half * 16 + j;
            unsigned int* slot = (unsigned int*)&Dls[wv][pos * 8];
#pragma unroll
            for (int h2 = 0; h2 < 4; ++h2) {
                unsigned int w = slot[h2];
                float p0 = __expf(bf2f((unsigned short)(w & 0xffff)) - mnew[2 * h2]);
                float p1 = __expf(bf2f((unsigned short)(w >> 16))    - mnew[2 * h2 + 1]);
                lblk[2 * h2]     += p0;
                lblk[2 * h2 + 1] += p1;
                slot[h2] = pack2(f2bf(p0), f2bf(p1));
            }
        }
#pragma unroll
        for (int h = 0; h < HH; ++h) {
            lblk[h] += __shfl_xor(lblk[h], 16, 32);
            l_run[h] = l_run[h] * alpha[h] + lblk[h];
            m_run[h] = mnew[h];
        }
        if (half == 0) {
#pragma unroll
            for (int h = 0; h < HH; ++h) Als[wv][qown][h] = alpha[h];
        }
        __syncthreads();

        // ---- stage 3: acc = acc*alpha + P @ V (bf16 WMMA, K = 32 keys) ----
#pragma unroll
        for (int h = 0; h < HH; ++h) {
            float ar[8];
#pragma unroll
            for (int r = 0; r < 8; ++r) ar[r] = Als[wv][r + 8 * half][h];
            v8u Au = {};
#pragma unroll
            for (int r = 0; r < 8; ++r) {
                int kk = a_kk(r, half);
                Au[r] = pack2(Dls[wv][(ln * 32 + kk) * 8 + h],
                              Dls[wv][(ln * 32 + kk + 1) * 8 + h]);
            }
#pragma unroll
            for (int dh = 0; dh < 2; ++dh) {
                const unsigned short* vrow =
                    Vt + ((size_t)m * 256 + h * 32 + dh * 16 + ln) * 1024 + kbase;
                v8u Bu = {};
#pragma unroll
                for (int r = 0; r < 8; ++r)
                    Bu[r] = *(const unsigned int*)(vrow + b_kk(r, half));
                v8f c;
#pragma unroll
                for (int r = 0; r < 8; ++r) c[r] = acc[h][dh][r] * ar[r];
                acc[h][dh] = wmma_bf16(Au, Bu, c);
            }
        }
        __syncthreads();  // LDS reused next iteration
    }

    // ---- epilogue: normalize by l and write O ----
    if (half == 0) {
#pragma unroll
        for (int h = 0; h < HH; ++h) Als[wv][qown][h] = 1.0f / l_run[h];
    }
    __syncthreads();
#pragma unroll
    for (int h = 0; h < HH; ++h) {
#pragma unroll
        for (int dh = 0; dh < 2; ++dh) {
#pragma unroll
            for (int r = 0; r < 8; ++r) {
                int qq = r + 8 * half;
                O[((size_t)m * NQC + q0 + qq) * 256 + h * 32 + dh * 16 + ln] =
                    acc[h][dh][r] * Als[wv][qq][h];
            }
        }
    }
}

// ---------------- kernel 3: output projection O @ Wout + bout ----------------

__global__ void out_gemm(const float* __restrict__ O, const float* __restrict__ Wout,
                         const float* __restrict__ bout, float* __restrict__ out) {
    int gid  = blockIdx.x * (blockDim.x >> 5) + (threadIdx.x >> 5);
    int lane = threadIdx.x & 31;
    int half = lane >> 4;
    int ln   = lane & 15;
    int row0 = (gid >> 4) * 16;
    int col0 = (gid & 15) * 16;

    v8f acc = {};
    for (int kk0 = 0; kk0 < 256; kk0 += 32) {
        v8u Au = {}, Bu = {};
#pragma unroll
        for (int r = 0; r < 8; ++r) {
            const float* pz = O + (size_t)(row0 + ln) * 256 + kk0 + a_kk(r, half);
            Au[r] = pack2(f2bf(pz[0]), f2bf(pz[1]));
            int kb = kk0 + b_kk(r, half);
            Bu[r] = pack2(f2bf(Wout[(size_t)kb * 256 + col0 + ln]),
                          f2bf(Wout[(size_t)(kb + 1) * 256 + col0 + ln]));
        }
        acc = wmma_bf16(Au, Bu, acc);
    }
    float bb = bout[col0 + ln];
#pragma unroll
    for (int r = 0; r < 8; ++r)
        out[(size_t)(row0 + r + 8 * half) * 256 + col0 + ln] = acc[r] + bb;
}

// ---------------- kernel 4: xq passthrough ----------------

__global__ void copy_xq(const float* __restrict__ src, float* __restrict__ dst) {
    int i = blockIdx.x * blockDim.x + threadIdx.x;
    if (i < MB * NQC * 2) dst[i] = src[i];
}

// ---------------- launch ----------------

extern "C" void kernel_launch(void* const* d_in, const int* in_sizes, int n_in,
                              void* d_out, int out_size, void* d_ws, size_t ws_size,
                              hipStream_t stream) {
    (void)in_sizes; (void)n_in; (void)out_size; (void)ws_size;

    const float* zq   = (const float*)d_in[0];
    const float* zk   = (const float*)d_in[1];
    const float* zv   = (const float*)d_in[2];
    const float* xq   = (const float*)d_in[3];
    const float* xkv  = (const float*)d_in[4];
    const float* Wq   = (const float*)d_in[5];
    const float* Wk   = (const float*)d_in[6];
    const float* Wv   = (const float*)d_in[7];
    const float* Wout = (const float*)d_in[8];
    const float* bout = (const float*)d_in[9];
    const float* W1   = (const float*)d_in[10];
    const float* b1   = (const float*)d_in[11];
    const float* W2   = (const float*)d_in[12];
    const float* b2   = (const float*)d_in[13];

    char* ws = (char*)d_ws;
    unsigned short* Qb = (unsigned short*)(ws);                              // 4 MB
    unsigned short* Kb = (unsigned short*)(ws + (size_t)4  * 1024 * 1024);   // 4 MB
    unsigned short* Vt = (unsigned short*)(ws + (size_t)8  * 1024 * 1024);   // 4 MB
    float*          O  = (float*)        (ws + (size_t)12 * 1024 * 1024);    // 8 MB
    float* out = (float*)d_out;

    // projections (bf16); V transposed to [m][inner][key]
    proj_gemm_n<<<1024, 256, 0, stream>>>(zq, Wq, Qb);
    proj_gemm_n<<<1024, 256, 0, stream>>>(zk, Wk, Kb);
    proj_gemm_t<<<1024, 256, 0, stream>>>(zv, Wv, Vt);

    // fused token-dots -> WMMA kernel-MLP -> softmax -> P@V
    attn_kernel<<<MB * (NQC / 32), 64, 0, stream>>>(Qb, Kb, Vt, xq, xkv,
                                                    W1, b1, W2, b2, O);

    // output projection + bias
    out_gemm<<<1024, 256, 0, stream>>>(O, Wout, bout, out);

    // xq passthrough appended after the main output
    copy_xq<<<64, 256, 0, stream>>>(xq, out + (size_t)MB * NQC * 256);
}
